// SelfAttentionLayer_10857677324873
// MI455X (gfx1250) — matrix operand order repaired and checked by reading, not compiled, not run on hardware
//
#include <hip/hip_runtime.h>
#include <hip/hip_bf16.h>

typedef unsigned short u16;
typedef __attribute__((ext_vector_type(16))) __bf16 v16bf;
typedef __attribute__((ext_vector_type(8)))  float  v8f;

union BfVec {
    v16bf v;
    uint4 q[2];
};

__device__ __forceinline__ u16 f2bf(float f) {
    unsigned u = __float_as_uint(f);
    unsigned r = u + 0x7fffu + ((u >> 16) & 1u);   // round-to-nearest-even
    return (u16)(r >> 16);
}

// low 32 bits of a flat pointer to __shared__ == LDS byte offset on amdgcn
__device__ __forceinline__ unsigned lds_off(const void* p) {
    return (unsigned)(uintptr_t)p;
}

// ---------------------------------------------------------------------------
// Weight prep: f32 [K x N] row-major -> bf16 WMMA-ready lane-major tiles.
//   out[t*512 + lane*16 + j] = B[kt*32 + (lane>=16 ? 16 : 0) + j, nt*16 + (lane&15)]
// in contiguous k order (k = 16*(lane/16) + j), the same order load_a_frag uses for A.
// ---------------------------------------------------------------------------
__global__ void prep_b_kernel(const float* __restrict__ B, u16* __restrict__ out,
                              int K, int N) {
    int idx = blockIdx.x * 256 + threadIdx.x;
    if (idx >= K * N) return;
    int t    = idx >> 9;
    int lane = (idx >> 4) & 31;
    int j    = idx & 15;
    int ntiles = N >> 4;
    int kt = t / ntiles, nt = t % ntiles;
    int k = kt * 32 + ((lane >> 4) << 4) + j;
    int n = nt * 16 + (lane & 15);
    out[idx] = f2bf(B[(size_t)k * N + n]);
}

// x f32 [M x 256] -> bf16 into cat[:, 0:256] (row stride 512)
__global__ void conv_x_kernel(const float* __restrict__ x, u16* __restrict__ cat) {
    size_t i = (size_t)blockIdx.x * 256 + threadIdx.x;
    size_t row = i >> 8;
    int col = (int)(i & 255);
    cat[row * 512 + col] = f2bf(x[i]);
}

// ---------------------------------------------------------------------------
// bf16 WMMA GEMM: block = 256 threads = 8 waves; wave computes 32x32 output via
// 2x2 tiles of v_wmma_f32_16x16x32_bf16.
// MODE: 0 = f32 out, 1 = elu(acc)+1 f32 out, 2 = acc*scale f32 out,
//       3 = relu(acc) -> bf16 out.
// ---------------------------------------------------------------------------
__device__ __forceinline__ v16bf load_a_frag(const u16* __restrict__ A, int lda,
                                             int row0, int kk, int lane) {
    int m    = row0 + (lane & 15);
    int base = kk + ((lane >> 4) << 4);           // lanes 16-31 take cols +16
    const u16* p = A + (size_t)m * lda + base;
    BfVec r;
    r.q[0] = *(const uint4*)(p);                   // K offsets 0..7
    r.q[1] = *(const uint4*)(p + 8);               // K offsets 8..15
    return r.v;
}

__device__ __forceinline__ v16bf load_b_frag(const u16* __restrict__ Bp,
                                             int tile, int lane) {
    const u16* p = Bp + ((size_t)tile << 9) + (lane << 4);
    BfVec r;
    r.q[0] = ((const uint4*)p)[0];
    r.q[1] = ((const uint4*)p)[1];
    return r.v;
}

template <int MODE>
__device__ __forceinline__ void store_tile(void* out, int ldo, int r0, int c0,
                                           v8f c, int lane, float scale) {
    int col   = c0 + (lane & 15);
    int rbase = r0 + ((lane >> 4) << 3);
#pragma unroll
    for (int i = 0; i < 8; ++i) {
        float v = c[i];
        if (MODE == 1) v = (v > 0.f) ? (v + 1.f) : __expf(v);  // elu(v)+1
        if (MODE == 2) v *= scale;
        size_t idx = (size_t)(rbase + i) * ldo + col;
        if (MODE == 3) {
            ((u16*)out)[idx] = f2bf(fmaxf(v, 0.f));
        } else {
            ((float*)out)[idx] = v;
        }
    }
}

template <int MODE>
__global__ __launch_bounds__(256) void gemm_bf16_kernel(
        const u16* __restrict__ A, int lda, const u16* __restrict__ Bp,
        int Ktot, int Ntot, void* __restrict__ out, float scale) {
    int lane = threadIdx.x & 31;
    int wave = threadIdx.x >> 5;
    int wm = wave & 3, wn = wave >> 2;
    int row0 = blockIdx.x * 128 + wm * 32;
    int col0 = blockIdx.y * 64 + wn * 32;
    int ntiles = Ntot >> 4;

    v8f c00 = {}, c01 = {}, c10 = {}, c11 = {};

    for (int kk = 0; kk < Ktot; kk += 32) {
        v16bf a0 = load_a_frag(A, lda, row0, kk, lane);
        v16bf a1 = load_a_frag(A, lda, row0 + 16, kk, lane);
        int tbase = (kk >> 5) * ntiles + (col0 >> 4);
        v16bf b0 = load_b_frag(Bp, tbase, lane);
        v16bf b1 = load_b_frag(Bp, tbase + 1, lane);
        c00 = __builtin_amdgcn_wmma_f32_16x16x32_bf16(false, a0, false, b0,
                                                      (short)0, c00, false, false);
        c01 = __builtin_amdgcn_wmma_f32_16x16x32_bf16(false, a0, false, b1,
                                                      (short)0, c01, false, false);
        c10 = __builtin_amdgcn_wmma_f32_16x16x32_bf16(false, a1, false, b0,
                                                      (short)0, c10, false, false);
        c11 = __builtin_amdgcn_wmma_f32_16x16x32_bf16(false, a1, false, b1,
                                                      (short)0, c11, false, false);
    }
    store_tile<MODE>(out, Ntot, row0,      col0,      c00, lane, scale);
    store_tile<MODE>(out, Ntot, row0,      col0 + 16, c01, lane, scale);
    store_tile<MODE>(out, Ntot, row0 + 16, col0,      c10, lane, scale);
    store_tile<MODE>(out, Ntot, row0 + 16, col0 + 16, c11, lane, scale);
}

// ---------------------------------------------------------------------------
// KV[n,h,d,dv] = sum_s K[n,s,h,d] * V[n,s,h,dv];  Ksum[n,h,d] = sum_s K.
// One block per (n,h); 1024 threads = (dv, d); s-tiles staged into LDS with
// CDNA5 async global->LDS copies (ASYNCcnt), no VGPR round-trip.
// ---------------------------------------------------------------------------
__global__ __launch_bounds__(1024) void kv_kernel(
        const float* __restrict__ Kf, const float* __restrict__ V,
        float* __restrict__ KV, float* __restrict__ Ksum, int L) {
    int n = blockIdx.x >> 3;
    int h = blockIdx.x & 7;
    int d  = threadIdx.x & 31;
    int dv = threadIdx.x >> 5;

    __shared__ float sk[32][32];
    __shared__ float sv[32][32];

    float acc = 0.f;
    float ks  = 0.f;
    int ls = threadIdx.x >> 5;   // s within tile for loading
    int lc = threadIdx.x & 31;   // channel for loading

    unsigned lk = lds_off(&sk[ls][lc]);
    unsigned lv = lds_off(&sv[ls][lc]);

    for (int s0 = 0; s0 < L; s0 += 32) {
        __syncthreads();
        size_t src = ((size_t)(n * L + s0 + ls) * 8 + h) * 32 + lc;
        const float* gk = Kf + src;
        const float* gv = V + src;
        // async global -> LDS (one b32 per lane per array)
        asm volatile("global_load_async_to_lds_b32 %0, %1, off"
                     :: "v"(lk), "v"(gk) : "memory");
        asm volatile("global_load_async_to_lds_b32 %0, %1, off"
                     :: "v"(lv), "v"(gv) : "memory");
        asm volatile("s_wait_asynccnt 0x0" ::: "memory");
        __syncthreads();
#pragma unroll
        for (int s = 0; s < 32; ++s) {
            float kv = sk[s][d];
            acc += kv * sv[s][dv];
            if (dv == 0) ks += kv;
        }
    }
    KV[(((size_t)n * 8 + h) * 32 + d) * 32 + dv] = acc;
    if (dv == 0) Ksum[((size_t)n * 8 + h) * 32 + d] = ks;
}

// ---------------------------------------------------------------------------
// msg[n,l,h,:] = (Q[n,l,h,:] @ KV[n,h]) * L / (Q·Ksum + eps)  -> bf16
// Block = 32 tokens x 8 heads (256 threads); 32KB KV panel staged into LDS
// with async b128 global->LDS copies.
// ---------------------------------------------------------------------------
__global__ __launch_bounds__(256) void msg_kernel(
        const float* __restrict__ Q, const float* __restrict__ KV,
        const float* __restrict__ Ksum, u16* __restrict__ msgpre,
        int L, float Lf) {
    __shared__ float sKV[8 * 32 * 32];
    __shared__ float sKs[8 * 32];

    int tid = threadIdx.x;
    int n    = blockIdx.x / (L / 32);
    int tok0 = (blockIdx.x % (L / 32)) * 32;

    const float* kvsrc = KV + (size_t)n * 8 * 32 * 32;
    {
        unsigned lbase = lds_off(&sKV[0]) + (unsigned)tid * 16u;
#pragma unroll
        for (int it = 0; it < 8; ++it) {
            const float* gp = kvsrc + it * 1024 + tid * 4;   // 4 floats = b128
            unsigned lp = lbase + (unsigned)it * 4096u;
            asm volatile("global_load_async_to_lds_b128 %0, %1, off"
                         :: "v"(lp), "v"(gp) : "memory");
        }
    }
    sKs[tid] = Ksum[(size_t)n * 256 + tid];
    asm volatile("s_wait_asynccnt 0x0" ::: "memory");
    __syncthreads();

    int h  = tid & 7;
    int lt = tid >> 3;
    int l  = tok0 + lt;

    const float* q = Q + ((size_t)(n * L + l) * 8 + h) * 32;
    float qr[32];
    float zden = 0.f;
#pragma unroll
    for (int d = 0; d < 32; ++d) {
        qr[d] = q[d];
        zden += qr[d] * sKs[h * 32 + d];
    }
    float z = Lf / (zden + 1e-6f);

    float acc[32];
#pragma unroll
    for (int dv = 0; dv < 32; ++dv) acc[dv] = 0.f;
    for (int d = 0; d < 32; ++d) {
        float qd = qr[d];
        const float* kvrow = &sKV[(h * 32 + d) * 32];
#pragma unroll
        for (int dv = 0; dv < 32; ++dv) acc[dv] += qd * kvrow[dv];
    }

    u16* o = msgpre + (size_t)(n * L + l) * 256 + h * 32;
#pragma unroll
    for (int dv = 0; dv < 32; ++dv) o[dv] = f2bf(acc[dv] * z);
}

// ---------------------------------------------------------------------------
// LayerNorm: one row (C=256) per wave; 8 waves per block.
// ---------------------------------------------------------------------------
__device__ __forceinline__ float wave_sum(float s) {
#pragma unroll
    for (int off = 16; off > 0; off >>= 1) s += __shfl_xor(s, off, 32);
    return s;
}

__global__ __launch_bounds__(256) void ln1_kernel(
        const float* __restrict__ t1, const float* __restrict__ g,
        const float* __restrict__ b, u16* __restrict__ cat) {
    int wave = threadIdx.x >> 5, lane = threadIdx.x & 31;
    size_t row = (size_t)blockIdx.x * 8 + wave;
    const float* p = t1 + row * 256 + lane * 8;
    float v[8], s = 0.f;
#pragma unroll
    for (int i = 0; i < 8; ++i) { v[i] = p[i]; s += v[i]; }
    float mu = wave_sum(s) * (1.f / 256.f);
    float vs = 0.f;
#pragma unroll
    for (int i = 0; i < 8; ++i) { float d = v[i] - mu; vs += d * d; }
    float rsq = rsqrtf(wave_sum(vs) * (1.f / 256.f) + 1e-5f);
    u16* o = cat + row * 512 + 256 + lane * 8;
#pragma unroll
    for (int i = 0; i < 8; ++i) {
        int c = lane * 8 + i;
        o[i] = f2bf((v[i] - mu) * rsq * g[c] + b[c]);
    }
}

__global__ __launch_bounds__(256) void ln2res_kernel(
        const float* __restrict__ t2, const float* __restrict__ g,
        const float* __restrict__ b, const float* __restrict__ x,
        float* __restrict__ out) {
    int wave = threadIdx.x >> 5, lane = threadIdx.x & 31;
    size_t row = (size_t)blockIdx.x * 8 + wave;
    const float* p = t2 + row * 256 + lane * 8;
    float v[8], s = 0.f;
#pragma unroll
    for (int i = 0; i < 8; ++i) { v[i] = p[i]; s += v[i]; }
    float mu = wave_sum(s) * (1.f / 256.f);
    float vs = 0.f;
#pragma unroll
    for (int i = 0; i < 8; ++i) { float d = v[i] - mu; vs += d * d; }
    float rsq = rsqrtf(wave_sum(vs) * (1.f / 256.f) + 1e-5f);
#pragma unroll
    for (int i = 0; i < 8; ++i) {
        int c = lane * 8 + i;
        size_t idx = row * 256 + c;
        out[idx] = x[idx] + (v[i] - mu) * rsq * g[c] + b[c];
    }
}

// ---------------------------------------------------------------------------
extern "C" void kernel_launch(void* const* d_in, const int* in_sizes, int n_in,
                              void* d_out, int out_size, void* d_ws, size_t ws_size,
                              hipStream_t stream) {
    const float* x  = (const float*)d_in[0];
    const float* Wq = (const float*)d_in[1];
    const float* Wk = (const float*)d_in[2];
    const float* Wv = (const float*)d_in[3];
    const float* Wm = (const float*)d_in[4];
    const float* W1 = (const float*)d_in[5];
    const float* W2 = (const float*)d_in[6];
    const float* g1 = (const float*)d_in[7];
    const float* b1 = (const float*)d_in[8];
    const float* g2 = (const float*)d_in[9];
    const float* b2 = (const float*)d_in[10];

    const int N = 4, L = 16384, H = 8;
    const size_t M = (size_t)N * L;               // 65536 tokens

    char* ws = (char*)d_ws;
    size_t off = 0;
    auto alloc = [&](size_t bytes) {
        size_t o = off;
        off += (bytes + 255) & ~(size_t)255;
        return o;
    };
    u16*   cat    = (u16*)  (ws + alloc(M * 512 * 2));     // [x | ln1(msg)] bf16
    float* Qb     = (float*)(ws + alloc(M * 256 * 4));
    float* Kb     = (float*)(ws + alloc(M * 256 * 4));
    float* Vb     = (float*)(ws + alloc(M * 256 * 4));
    u16*   msgpre = (u16*)  (ws + alloc(M * 256 * 2));
    float* KV     = (float*)(ws + alloc((size_t)N * H * 32 * 32 * 4));
    float* Ksum   = (float*)(ws + alloc((size_t)N * H * 32 * 4));
    u16* Wqp = (u16*)(ws + alloc(256 * 256 * 2));
    u16* Wkp = (u16*)(ws + alloc(256 * 256 * 2));
    u16* Wvp = (u16*)(ws + alloc(256 * 256 * 2));
    u16* Wmp = (u16*)(ws + alloc(256 * 256 * 2));
    u16* W1p = (u16*)(ws + alloc(512 * 512 * 2));
    u16* W2p = (u16*)(ws + alloc(512 * 256 * 2));
    // buffer reuse after attention core:
    float* t1   = Qb;          // msgpre @ Wm
    u16*   hbuf = (u16*)Kb;    // relu(cat @ W1) bf16 [M x 512]
    float* t2   = Vb;          // hbuf @ W2

    // --- weight prep ---
    prep_b_kernel<<<(256 * 256 + 255) / 256, 256, 0, stream>>>(Wq, Wqp, 256, 256);
    prep_b_kernel<<<(256 * 256 + 255) / 256, 256, 0, stream>>>(Wk, Wkp, 256, 256);
    prep_b_kernel<<<(256 * 256 + 255) / 256, 256, 0, stream>>>(Wv, Wvp, 256, 256);
    prep_b_kernel<<<(256 * 256 + 255) / 256, 256, 0, stream>>>(Wm, Wmp, 256, 256);
    prep_b_kernel<<<(512 * 512 + 255) / 256, 256, 0, stream>>>(W1, W1p, 512, 512);
    prep_b_kernel<<<(512 * 256 + 255) / 256, 256, 0, stream>>>(W2, W2p, 512, 256);

    // --- x -> bf16 into cat[:, :256] ---
    conv_x_kernel<<<(unsigned)(M * 256 / 256), 256, 0, stream>>>(x, cat);

    // --- QKV projections (WMMA) with fused feature-map epilogues ---
    dim3 g256((unsigned)(M / 128), 256 / 64);
    gemm_bf16_kernel<1><<<g256, 256, 0, stream>>>(cat, 512, Wqp, 256, 256, Qb, 0.f);
    gemm_bf16_kernel<1><<<g256, 256, 0, stream>>>(cat, 512, Wkp, 256, 256, Kb, 0.f);
    gemm_bf16_kernel<2><<<g256, 256, 0, stream>>>(cat, 512, Wvp, 256, 256, Vb,
                                                  1.0f / (float)L);

    // --- linear-attention core ---
    kv_kernel<<<N * H, 1024, 0, stream>>>(Kb, Vb, KV, Ksum, L);
    msg_kernel<<<(unsigned)(M / 32), 256, 0, stream>>>(Qb, KV, Ksum, msgpre, L,
                                                       (float)L);

    // --- merge heads: msgpre @ Wm (WMMA) ---
    gemm_bf16_kernel<0><<<g256, 256, 0, stream>>>(msgpre, 256, Wmp, 256, 256, t1, 0.f);

    // --- LN1 -> cat[:, 256:512] bf16 ---
    ln1_kernel<<<(unsigned)(M / 8), 256, 0, stream>>>(t1, g1, b1, cat);

    // --- MLP: relu(cat @ W1) -> bf16, then @ W2 -> f32 (WMMA) ---
    dim3 g512((unsigned)(M / 128), 512 / 64);
    gemm_bf16_kernel<3><<<g512, 256, 0, stream>>>(cat, 512, W1p, 512, 512, hbuf, 0.f);
    gemm_bf16_kernel<0><<<g256, 256, 0, stream>>>(hbuf, 512, W2p, 512, 256, t2, 0.f);

    // --- LN2 + residual -> out ---
    ln2res_kernel<<<(unsigned)(M / 8), 256, 0, stream>>>(t2, g2, b2, x,
                                                         (float*)d_out);
}
